// DeepOHeat_ST_4011499454816
// MI455X (gfx1250) — matrix operand order repaired and checked by reading, not compiled, not run on hardware
//
#include <hip/hip_runtime.h>
#include <hip/hip_bf16.h>

typedef __attribute__((ext_vector_type(16))) _Float16 v16h;
typedef __attribute__((ext_vector_type(8)))  float    v8f;

// ---------------------------------------------------------------------------
// Phase 1a: Fourier features  feat = [cos(x*B), sin(x*B)]  (per trunk)
// ---------------------------------------------------------------------------
__global__ void feat_kernel(const float* __restrict__ x1, const float* __restrict__ x2,
                            const float* __restrict__ x3, const float* __restrict__ x4,
                            const float* __restrict__ Bm, float* __restrict__ feat)
{
    const int net = blockIdx.x;
    const float* xs[4] = {x1, x2, x3, x4};
    const float* x = xs[net];
    const int N = (net == 3) ? 32 : 64;
    float* fo = feat + net * 64 * 128;
    for (int idx = threadIdx.x; idx < N * 64; idx += blockDim.x) {
        int row = idx >> 6, c = idx & 63;
        float p = x[row] * Bm[c];
        fo[row * 128 + c]      = cosf(p);
        fo[row * 128 + 64 + c] = sinf(p);
    }
}

// ---------------------------------------------------------------------------
// Phase 1b: generic linear layer  out = act(in @ W^T + b)
// blockIdx.y = network index (stacked trunk weights); act=1 -> swish
// ---------------------------------------------------------------------------
__global__ void lin_kernel(const float* __restrict__ in, const float* __restrict__ W,
                           const float* __restrict__ bias, float* __restrict__ out,
                           int Nrows, int K, int M, int act, int trunkmode,
                           int in_ns, int w_ns, int b_ns, int out_ns)
{
    const int net = blockIdx.y;
    const int N = trunkmode ? ((net == 3) ? 32 : 64) : Nrows;
    const int idx = blockIdx.x * blockDim.x + threadIdx.x;
    if (idx >= N * M) return;
    const int row = idx / M;
    const int m   = idx - row * M;
    const float* ip = in + net * in_ns + row * K;
    const float* wp = W  + net * w_ns  + m * K;
    float acc = bias[net * b_ns + m];
    #pragma unroll 4
    for (int k = 0; k < K; k += 4) {
        acc += ip[k]     * wp[k];
        acc += ip[k + 1] * wp[k + 1];
        acc += ip[k + 2] * wp[k + 2];
        acc += ip[k + 3] * wp[k + 3];
    }
    if (act) acc = acc / (1.0f + expf(-acc));           // swish
    out[net * out_ns + row * M + m] = acc;
}

// ---------------------------------------------------------------------------
// Phase 2: rank-64 CP reconstruction via WMMA f16->f32.
// One wave handles (b, i, j0..j0+3); per j: out[k(64), l(32)] =
//   sum_z t3[k,z] * ( t1[i,z]*t2[j,z]*bb[b,z]*t4[l,z] )
// A = t3 (shared, converted once per wave), B rebuilt per j in f16.
// ---------------------------------------------------------------------------
__global__ __launch_bounds__(256) void cp_einsum_kernel(
    const float* __restrict__ tbuf, const float* __restrict__ bb,
    float* __restrict__ out)
{
    const int lane = threadIdx.x & 31;
    const int wave = blockIdx.x * (blockDim.x >> 5) + (threadIdx.x >> 5);
    const int jblk = wave & 15;          // 16 j-blocks of 4
    const int i    = (wave >> 4) & 63;
    const int b    = wave >> 10;
    const int hi   = lane >> 4;          // lane half (0/1)
    const int nlo  = lane & 15;

    const float* t1 = tbuf;              // (64,64) i-factor
    const float* t2 = tbuf + 4096;       // (64,64) j-factor
    const float* t3 = tbuf + 8192;       // (64,64) k-factor -> A matrix
    const float* t4 = tbuf + 12288;      // (32,64) l-factor

    // --- A fragments from t3 (16x32 f16 per WMMA; contiguous k order) -------------
    // half h : z = s*32 + hi*16 + h
    v16h afrag[4][2];
    #pragma unroll
    for (int m = 0; m < 4; ++m) {
        const float* row = t3 + (m * 16 + nlo) * 64;
        #pragma unroll
        for (int s = 0; s < 2; ++s) {
            const int zb = s * 32 + hi * 16;
            #pragma unroll
            for (int h = 0; h < 8; ++h) {
                afrag[m][s][h]     = (_Float16)row[zb + h];
                afrag[m][s][h + 8] = (_Float16)row[zb + 8 + h];
            }
        }
    }

    // --- per-lane j-invariant products: v[s][n][h] = t1*bb*t4 -------------
    // B-layout z for half h: z = s*32 + hi*16 + h ; column l = n*16 + nlo
    float vv[2][2][16];
    #pragma unroll
    for (int s = 0; s < 2; ++s) {
        const int zb = s * 32 + hi * 16;
        const float* r1 = t1 + i * 64 + zb;
        const float* rb = bb + b * 64 + zb;
        #pragma unroll
        for (int n = 0; n < 2; ++n) {
            const float* r4 = t4 + (n * 16 + nlo) * 64 + zb;
            #pragma unroll
            for (int h = 0; h < 16; ++h)
                vv[s][n][h] = r1[h] * rb[h] * r4[h];
        }
    }

    float* outb = out + ((size_t)((b * 64 + i) * 64 + jblk * 4)) * 2048;

    for (int jj = 0; jj < 4; ++jj) {
        const int j = jblk * 4 + jj;
        v8f acc[4][2];
        #pragma unroll
        for (int m = 0; m < 4; ++m)
            #pragma unroll
            for (int n = 0; n < 2; ++n)
                #pragma unroll
                for (int r = 0; r < 8; ++r) acc[m][n][r] = 0.0f;

        #pragma unroll
        for (int s = 0; s < 2; ++s) {
            const int zb = s * 32 + hi * 16;
            const float* r2 = t2 + j * 64 + zb;
            float t2v[16];
            #pragma unroll
            for (int h = 0; h < 16; ++h) t2v[h] = r2[h];

            v16h bf[2];
            #pragma unroll
            for (int n = 0; n < 2; ++n)
                #pragma unroll
                for (int h = 0; h < 16; ++h)
                    bf[n][h] = (_Float16)(vv[s][n][h] * t2v[h]);

            #pragma unroll
            for (int m = 0; m < 4; ++m)
                #pragma unroll
                for (int n = 0; n < 2; ++n)
                    acc[m][n] = __builtin_amdgcn_wmma_f32_16x16x32_f16(
                        false, afrag[m][s], false, bf[n],
                        (short)0, acc[m][n], false, false);
        }

        // store: C VGPR r -> row M = m*16 + r + hi*8, col N = n*16 + nlo
        float* oj = outb + (size_t)jj * 2048;
        #pragma unroll
        for (int m = 0; m < 4; ++m)
            #pragma unroll
            for (int n = 0; n < 2; ++n)
                #pragma unroll
                for (int r = 0; r < 8; ++r)
                    oj[(m * 16 + r + hi * 8) * 32 + n * 16 + nlo] = acc[m][n][r];
    }
}

// ---------------------------------------------------------------------------
extern "C" void kernel_launch(void* const* d_in, const int* in_sizes, int n_in,
                              void* d_out, int out_size, void* d_ws, size_t ws_size,
                              hipStream_t stream)
{
    const float* x1  = (const float*)d_in[0];
    const float* x2  = (const float*)d_in[1];
    const float* x3  = (const float*)d_in[2];
    const float* x4  = (const float*)d_in[3];
    const float* f   = (const float*)d_in[4];
    const float* Bm  = (const float*)d_in[5];
    const float* tW0 = (const float*)d_in[6];
    const float* tb0 = (const float*)d_in[7];
    const float* tW1 = (const float*)d_in[8];
    const float* tb1 = (const float*)d_in[9];
    const float* tW2 = (const float*)d_in[10];
    const float* tb2 = (const float*)d_in[11];
    const float* tW3 = (const float*)d_in[12];
    const float* tb3 = (const float*)d_in[13];
    const float* bW0 = (const float*)d_in[14];
    const float* bb0 = (const float*)d_in[15];
    const float* bW1 = (const float*)d_in[16];
    const float* bb1 = (const float*)d_in[17];
    const float* bW2 = (const float*)d_in[18];
    const float* bb2 = (const float*)d_in[19];
    const float* bW3 = (const float*)d_in[20];
    const float* bb3 = (const float*)d_in[21];

    float* ws   = (float*)d_ws;
    float* feat = ws;               // 4*64*128 = 32768
    float* hA   = ws + 32768;       // 4*64*256 = 65536
    float* hB   = ws + 98304;       // 65536
    float* t    = ws + 163840;      // 4*64*64  = 16384 (t1|t2|t3|t4 stacked)
    float* bh1  = ws + 180224;      // 8*256
    float* bh2  = ws + 182272;      // 8*256
    float* bbv  = ws + 184320;      // 8*64

    // trunk features
    feat_kernel<<<4, 256, 0, stream>>>(x1, x2, x3, x4, Bm, feat);

    // trunk MLPs (grid.y = 4 nets)
    lin_kernel<<<dim3(64, 4), 256, 0, stream>>>(feat, tW0, tb0, hA, 64, 128, 256, 1, 1,
                                                8192, 32768, 256, 16384);
    lin_kernel<<<dim3(64, 4), 256, 0, stream>>>(hA, tW1, tb1, hB, 64, 256, 256, 1, 1,
                                                16384, 65536, 256, 16384);
    lin_kernel<<<dim3(64, 4), 256, 0, stream>>>(hB, tW2, tb2, hA, 64, 256, 256, 1, 1,
                                                16384, 65536, 256, 16384);
    lin_kernel<<<dim3(16, 4), 256, 0, stream>>>(hA, tW3, tb3, t, 64, 256, 64, 0, 1,
                                                16384, 16384, 64, 4096);

    // branch MLP
    lin_kernel<<<dim3(8, 1), 256, 0, stream>>>(f,   bW0, bb0, bh1, 8, 256, 256, 1, 0, 0, 0, 0, 0);
    lin_kernel<<<dim3(8, 1), 256, 0, stream>>>(bh1, bW1, bb1, bh2, 8, 256, 256, 1, 0, 0, 0, 0, 0);
    lin_kernel<<<dim3(8, 1), 256, 0, stream>>>(bh2, bW2, bb2, bh1, 8, 256, 256, 1, 0, 0, 0, 0, 0);
    lin_kernel<<<dim3(2, 1), 256, 0, stream>>>(bh1, bW3, bb3, bbv, 8, 256, 64, 0, 0, 0, 0, 0, 0);

    // CP einsum: 8192 waves (1024 blocks x 8 waves), 64 WMMA each
    cp_einsum_kernel<<<1024, 256, 0, stream>>>(t, bbv, (float*)d_out);
}
